// AttentionDecoderGRU_modified_33423435497666
// MI455X (gfx1250) — matrix operand order repaired and checked by reading, not compiled, not run on hardware
//
#include <hip/hip_runtime.h>
#include <cmath>

typedef _Float16 f16;
typedef __attribute__((ext_vector_type(16))) _Float16 v16h;
typedef __attribute__((ext_vector_type(8)))  float    v8f;

#define NHEADS 8
#define HD     96
#define BSZ    128
#define TLEN   20
#define LNUM   4
#define HDIM   768
#define IDIM   768
#define ODIM   512
#define NPAT   200

// ---------------------------------------------------------------------------
// Generic batched f16 WMMA GEMM:  C[M,N] = A[M,K] * op(B) + bias + res
//   BT=1 : B is [N,K] row-major (x @ W^T, "NT").  All NT shapes have N%128==0
//          -> no guards; A and B tiles staged with GLOBAL_LOAD_ASYNC_TO_LDS.
//   BT=0 : B is [K,N] row-major (probs @ V, "NN"); N may be 96 -> guarded,
//          transposed synchronous staging for B, async staging for A.
// Block: 256 threads (8 wave32), tile 128(M) x 128(N), K-chunk 32.
// Waves arranged 4(M) x 2(N); each wave: 32x64 = 2x4 WMMA tiles
// (8 x v_wmma_f32_16x16x32_f16 per wave per K-chunk).
// LDS is double-buffered: async copies of tile k+1 overlap WMMA of tile k;
// steady-state drain is s_wait_asynccnt <= (async ops of one tile per wave),
// relying on in-order per-wave async completion (ISA 08 sec 4.1).
// ---------------------------------------------------------------------------
struct GemmP {
  const f16* A;  int lda;  long bsA;
  const f16* B;  int ldb;  long bsB;
  const float* bias;
  const float* res; int ldr; long bsR;
  float* C32; f16* C16; int ldc; long bsC;
  int M, N, K;
};

#define BM 128
#define BN 128
#define BK 32
#define LDS_K 40   // row stride (halves): 32 data + 8 pad = 80B, 16B-aligned

__device__ __forceinline__ void async_b128(unsigned ldsByteOff, const f16* src) {
  asm volatile("global_load_async_to_lds_b128 %0, %1, off"
               :: "v"(ldsByteOff), "v"((unsigned long long)(uintptr_t)src)
               : "memory");
}

template <int BT>
__global__ __launch_bounds__(256) void gemm_f16_wmma(GemmP p) {
  // Single shared object => group-segment offset 0; LDS byte addresses for the
  // async-copy instruction are computed relative to &sm.
  __shared__ __align__(16) struct SMem {
    f16 As[2][BM][LDS_K];
    f16 Bs[2][BN][LDS_K];
  } sm;

  const int bz   = blockIdx.z;
  const f16* A   = p.A + (long)bz * p.bsA;
  const f16* B   = p.B + (long)bz * p.bsB;
  const int tid  = threadIdx.x;
  const int lane = tid & 31;
  const int w    = tid >> 5;
  const int wm   = w >> 1;       // 0..3  -> M offset wm*32
  const int wn   = w & 1;        // 0..1  -> N offset wn*64
  const int m0   = blockIdx.y * BM;
  const int n0   = blockIdx.x * BN;
  const int l16  = lane & 15;
  const int hi   = lane >> 4;    // 0 or 1

  v8f acc[2][4];
#pragma unroll
  for (int i = 0; i < 2; ++i)
#pragma unroll
    for (int j = 0; j < 4; ++j)
#pragma unroll
      for (int e = 0; e < 8; ++e) acc[i][j][e] = 0.0f;

  // Per-thread staging coordinates: 16 halves (32B) each, 128x32 tile total.
  const int sr = tid >> 1;              // 0..127 (row: A->m, B(NT)->n)
  const int sk = (tid & 1) * 16;        // 0 or 16 (k)
  const unsigned aDst[2] = {
      (unsigned)((char*)&sm.As[0][sr][sk] - (char*)&sm),
      (unsigned)((char*)&sm.As[1][sr][sk] - (char*)&sm)};
  const unsigned bDst[2] = {
      (unsigned)((char*)&sm.Bs[0][sr][sk] - (char*)&sm),
      (unsigned)((char*)&sm.Bs[1][sr][sk] - (char*)&sm)};

  const int nk = p.K / BK;

  // ---- tile staging (tile index kt -> k0 = kt*BK, into LDS buffer `buf`)
  auto stage = [&](int kt, int buf) {
    const int k0 = kt * BK;
    {
      const f16* src = A + (long)(m0 + sr) * p.lda + k0 + sk;
      async_b128(aDst[buf], src);
      async_b128(aDst[buf] + 16, src + 8);
    }
    if (BT) {
      const f16* src = B + (long)(n0 + sr) * p.ldb + k0 + sk;
      async_b128(bDst[buf], src);
      async_b128(bDst[buf] + 16, src + 8);
    } else {
      // B[k][n] -> transpose into Bs[buf][n][k] with N-guard (N can be 96)
      int kr = tid >> 3;                // 0..31 (k)
      int nc = (tid & 7) * 16;          // 0..112 (n)
      union { ulonglong2 q[2]; f16 h[16]; } u;
      if (n0 + nc + 16 <= p.N) {
        const f16* src = B + (long)(k0 + kr) * p.ldb + n0 + nc;
        u.q[0] = *(const ulonglong2*)src;
        u.q[1] = *(const ulonglong2*)(src + 8);
      } else {
#pragma unroll
        for (int j = 0; j < 16; ++j)
          u.h[j] = (n0 + nc + j < p.N) ? B[(long)(k0 + kr) * p.ldb + n0 + nc + j]
                                       : (f16)0;
      }
#pragma unroll
      for (int j = 0; j < 16; ++j) sm.Bs[buf][nc + j][kr] = u.h[j];
    }
  };

  stage(0, 0);  // prologue

  for (int i = 0; i < nk; ++i) {
    const int cur = i & 1;
    if (i + 1 < nk) {
      stage(i + 1, cur ^ 1);
      // Drain only tile i's async ops (in-order per-wave completion):
      // NT issues 4 async ops per wave per tile, NN issues 2 (A only).
      if (BT) asm volatile("s_wait_asynccnt 0x4" ::: "memory");
      else    asm volatile("s_wait_asynccnt 0x2" ::: "memory");
    } else {
      asm volatile("s_wait_asynccnt 0x0" ::: "memory");
    }
    __syncthreads();   // tile i visible to all waves

    // ---- build fragments per ISA VGPR layouts and issue WMMA
    v16h afrag[2], bfrag[4];
    {
      const int abase = hi * 8;         // A: lanes>=16 hold K+8
#pragma unroll
      for (int mt = 0; mt < 2; ++mt) {
        int m = wm * 32 + mt * 16 + l16;
        union { v16h v; f16 h[16]; } u;
#pragma unroll
        for (int i2 = 0; i2 < 8; ++i2) u.h[i2]     = sm.As[cur][m][abase + i2];
#pragma unroll
        for (int i2 = 0; i2 < 8; ++i2) u.h[8 + i2] = sm.As[cur][m][16 + abase + i2];
        afrag[mt] = u.v;
      }
      const int bbase = hi * 8;         // B: lanes>=16 hold K+8
#pragma unroll
      for (int nt = 0; nt < 4; ++nt) {
        int n = wn * 64 + nt * 16 + l16;
        union { v16h v; f16 h[16]; } u;
#pragma unroll
        for (int i2 = 0; i2 < 8; ++i2) u.h[i2]     = sm.Bs[cur][n][bbase + i2];
#pragma unroll
        for (int i2 = 0; i2 < 8; ++i2) u.h[8 + i2] = sm.Bs[cur][n][16 + bbase + i2];
        bfrag[nt] = u.v;
      }
    }
#pragma unroll
    for (int mt = 0; mt < 2; ++mt)
#pragma unroll
      for (int nt = 0; nt < 4; ++nt)
        acc[mt][nt] = __builtin_amdgcn_wmma_f32_16x16x32_f16(
            false, afrag[mt], false, bfrag[nt], (short)0, acc[mt][nt],
            false, false);
    __syncthreads();   // all waves done reading buffer `cur` (reused at i+2)
  }

  // ---- epilogue: bias + residual, f32/f16 stores; C layout per ISA
  const float* resB = p.res ? p.res + (long)bz * p.bsR : nullptr;
  float* C32 = p.C32 ? p.C32 + (long)bz * p.bsC : nullptr;
  f16*   C16 = p.C16 ? p.C16 + (long)bz * p.bsC : nullptr;
#pragma unroll
  for (int mt = 0; mt < 2; ++mt) {
#pragma unroll
    for (int nt = 0; nt < 4; ++nt) {
      int n = n0 + wn * 64 + nt * 16 + l16;
      if (!BT && n >= p.N) continue;
      int mb = m0 + wm * 32 + mt * 16 + hi * 8;
      float bv = p.bias ? p.bias[n] : 0.0f;
#pragma unroll
      for (int r = 0; r < 8; ++r) {
        int m = mb + r;
        float v = acc[mt][nt][r] + bv;
        if (resB) v += resB[(long)m * p.ldr + n];
        if (C32) C32[(long)m * p.ldc + n] = v;
        if (C16) C16[(long)m * p.ldc + n] = (f16)v;
      }
    }
  }
}

// ---------------------------------------------------------------------------
// Elementwise / reduction helper kernels
// ---------------------------------------------------------------------------
__global__ void f32_to_f16_kernel(const float* src, f16* dst, long n) {
  long i = (long)blockIdx.x * blockDim.x + threadIdx.x;
  if (i < n) dst[i] = (f16)src[i];
}

__global__ void zero_state_kernel(float* h32, f16* h16, long n) {
  long i = (long)blockIdx.x * blockDim.x + threadIdx.x;
  if (i < n) { h32[i] = 0.0f; h16[i] = (f16)0; }
}

// L2-normalize each patch row over I, then mean-pool groups of N/T rows.
// One block per (l, b, t); 256 threads, 3 elems/thread (I=768).
__global__ __launch_bounds__(256) void pool_norm_kernel(const float* pf, f16* pooled) {
  __shared__ float red[256];
  int blk = blockIdx.x;
  int t = blk % TLEN;
  int b = (blk / TLEN) % BSZ;
  int l = blk / (TLEN * BSZ);
  const int G = NPAT / TLEN;  // 10
  const float* base = pf + (((long)l * BSZ + b) * NPAT + (long)t * G) * IDIM;
  int tid = threadIdx.x;
  float a0 = 0.f, a1 = 0.f, a2 = 0.f;
  for (int j = 0; j < G; ++j) {
    const float* row = base + (long)j * IDIM;
    float v0 = row[tid], v1 = row[tid + 256], v2 = row[tid + 512];
    red[tid] = v0 * v0 + v1 * v1 + v2 * v2;
    __syncthreads();
    for (int off = 128; off > 0; off >>= 1) {
      if (tid < off) red[tid] += red[tid + off];
      __syncthreads();
    }
    float rn = rsqrtf(red[0]);
    __syncthreads();
    a0 += v0 * rn; a1 += v1 * rn; a2 += v2 * rn;
  }
  f16* out = pooled + ((long)(l * TLEN + t) * BSZ + b) * IDIM;
  const float invG = 1.0f / G;
  out[tid]       = (f16)(a0 * invG);
  out[tid + 256] = (f16)(a1 * invG);
  out[tid + 512] = (f16)(a2 * invG);
}

// GRU gates: h_new = (1-z)*n + z*h_prev ; gi/gh are [B, 3H] (r,z,n stacked)
__global__ void gru_gates_kernel(const float* gi, const float* gh,
                                 const float* hPrev, float* h32, f16* h16) {
  long idx = (long)blockIdx.x * blockDim.x + threadIdx.x;
  if (idx >= (long)BSZ * HDIM) return;
  int b = (int)(idx / HDIM), c = (int)(idx % HDIM);
  long base = (long)b * (3 * HDIM);
  float ir = gi[base + c], iz = gi[base + HDIM + c], in = gi[base + 2 * HDIM + c];
  float hr = gh[base + c], hz = gh[base + HDIM + c], hn = gh[base + 2 * HDIM + c];
  float r = 1.0f / (1.0f + __expf(-(ir + hr)));
  float z = 1.0f / (1.0f + __expf(-(iz + hz)));
  float n = tanhf(in + r * hn);
  float h = (1.0f - z) * n + z * hPrev[idx];
  h32[idx] = h;
  h16[idx] = (f16)h;
}

// Softmax over rows of length 128 with logit scale; one block (128 thr) per row.
__global__ __launch_bounds__(128) void softmax_rows_kernel(const float* S, f16* P,
                                                           float scale) {
  __shared__ float red[128];
  long row = blockIdx.x;
  int t = threadIdx.x;
  float v = S[row * 128 + t] * scale;
  red[t] = v; __syncthreads();
  for (int off = 64; off > 0; off >>= 1) {
    if (t < off) red[t] = fmaxf(red[t], red[t + off]);
    __syncthreads();
  }
  float mx = red[0]; __syncthreads();
  float e = __expf(v - mx);
  red[t] = e; __syncthreads();
  for (int off = 64; off > 0; off >>= 1) {
    if (t < off) red[t] += red[t + off];
    __syncthreads();
  }
  P[row * 128 + t] = (f16)(e / red[0]);
}

// LayerNorm over 768 cols; one block (256 thr) per row; writes f32 (opt) + f16.
__global__ __launch_bounds__(256) void layernorm_rows_kernel(const float* X,
                                                             const float* g,
                                                             const float* bta,
                                                             float* Y32, f16* Y16) {
  __shared__ float red[256];
  long row = blockIdx.x;
  int t = threadIdx.x;
  const float* x = X + row * HDIM;
  float s = x[t] + x[t + 256] + x[t + 512];
  red[t] = s; __syncthreads();
  for (int off = 128; off > 0; off >>= 1) {
    if (t < off) red[t] += red[t + off];
    __syncthreads();
  }
  float mean = red[0] / HDIM; __syncthreads();
  float d0 = x[t] - mean, d1 = x[t + 256] - mean, d2 = x[t + 512] - mean;
  red[t] = d0 * d0 + d1 * d1 + d2 * d2; __syncthreads();
  for (int off = 128; off > 0; off >>= 1) {
    if (t < off) red[t] += red[t + off];
    __syncthreads();
  }
  float inv = rsqrtf(red[0] / HDIM + 1e-5f);
#pragma unroll
  for (int k = 0; k < 3; ++k) {
    int c = t + k * 256;
    float y = (x[c] - mean) * inv * g[c] + bta[c];
    if (Y32) Y32[row * HDIM + c] = y;
    Y16[row * HDIM + c] = (f16)y;
  }
}

// out[b,t,o] = sum_l proj[l,t,b,o]*fus_w[l] + fus_b
__global__ void fuse_out_kernel(const float* proj, const float* fw,
                                const float* fb, float* out) {
  long i = (long)blockIdx.x * blockDim.x + threadIdx.x;
  long total = (long)BSZ * TLEN * ODIM;
  if (i >= total) return;
  int o = (int)(i % ODIM);
  long r = i / ODIM;
  int t = (int)(r % TLEN);
  int b = (int)(r / TLEN);
  const long sl = (long)TLEN * BSZ * ODIM;
  float s = fb[0];
#pragma unroll
  for (int l = 0; l < LNUM; ++l)
    s += proj[l * sl + ((long)t * BSZ + b) * ODIM + o] * fw[l];
  out[((long)b * TLEN + t) * ODIM + o] = s;
}

// ---------------------------------------------------------------------------
// Host-side launch plumbing
// ---------------------------------------------------------------------------
static void launch_gemm(hipStream_t st,
                        const f16* A, int lda, long bsA,
                        const f16* B, int ldb, long bsB, int bT,
                        const float* bias,
                        const float* res, int ldr, long bsR,
                        float* C32, f16* C16, int ldc, long bsC,
                        int M, int N, int K, int batch) {
  GemmP p;
  p.A = A; p.lda = lda; p.bsA = bsA;
  p.B = B; p.ldb = ldb; p.bsB = bsB;
  p.bias = bias;
  p.res = res; p.ldr = ldr; p.bsR = bsR;
  p.C32 = C32; p.C16 = C16; p.ldc = ldc; p.bsC = bsC;
  p.M = M; p.N = N; p.K = K;
  dim3 grid((N + BN - 1) / BN, (M + BM - 1) / BM, batch);
  if (bT) hipLaunchKernelGGL((gemm_f16_wmma<1>), grid, dim3(256), 0, st, p);
  else    hipLaunchKernelGGL((gemm_f16_wmma<0>), grid, dim3(256), 0, st, p);
}

extern "C" void kernel_launch(void* const* d_in, const int* in_sizes, int n_in,
                              void* d_out, int out_size, void* d_ws, size_t ws_size,
                              hipStream_t stream) {
  (void)in_sizes; (void)n_in; (void)out_size; (void)ws_size;
  const float* decoder_input = (const float*)d_in[0];
  const float* patch_features = (const float*)d_in[1];
  const float* gru_Wih = (const float*)d_in[2];
  const float* gru_Whh = (const float*)d_in[3];
  const float* gru_bih = (const float*)d_in[4];
  const float* gru_bhh = (const float*)d_in[5];
  const float* enc_W   = (const float*)d_in[6];
  const float* enc_b   = (const float*)d_in[7];
  const float* sa_in_w = (const float*)d_in[8];
  const float* sa_in_b = (const float*)d_in[9];
  const float* sa_out_w = (const float*)d_in[10];
  const float* sa_out_b = (const float*)d_in[11];
  const float* sa_fc_w = (const float*)d_in[12];
  const float* sa_fc_b = (const float*)d_in[13];
  const float* ln_g = (const float*)d_in[14];
  const float* ln_b = (const float*)d_in[15];
  const float* ca_in_w = (const float*)d_in[16];
  const float* ca_in_b = (const float*)d_in[17];
  const float* ca_out_w = (const float*)d_in[18];
  const float* ca_out_b = (const float*)d_in[19];
  const float* fc_w = (const float*)d_in[20];
  const float* fc_b = (const float*)d_in[21];
  const float* outp_w = (const float*)d_in[22];
  const float* outp_b = (const float*)d_in[23];
  const float* fus_w = (const float*)d_in[24];
  const float* fus_b = (const float*)d_in[25];

  const long LTB = (long)LNUM * TLEN * BSZ;  // 10240

  // Workspace allocator (256B aligned slices)
  char* wp = (char*)d_ws;
  auto a16 = [&](long n) { f16* p = (f16*)wp;   wp += ((n * 2 + 255) & ~255L); return p; };
  auto a32 = [&](long n) { float* p = (float*)wp; wp += ((n * 4 + 255) & ~255L); return p; };

  // f16 weights
  f16* wWih  = a16((long)3 * HDIM * IDIM);
  f16* wWhh  = a16((long)3 * HDIM * HDIM);
  f16* wEnc  = a16((long)HDIM * IDIM);
  f16* wSaIn = a16((long)3 * HDIM * HDIM);
  f16* wSaOut = a16((long)HDIM * HDIM);
  f16* wSaFc = a16((long)HDIM * HDIM);
  f16* wCaIn = a16((long)3 * HDIM * HDIM);
  f16* wCaOut = a16((long)HDIM * HDIM);
  f16* wFc   = a16((long)IDIM * HDIM);
  f16* wOutp = a16((long)ODIM * IDIM);
  // activations
  f16* dec0   = a16((long)BSZ * IDIM);
  f16* pooled = a16(LTB * IDIM);     // pooled -> (reused) enc2
  f16* encTmp = a16(LTB * HDIM);     // enc1
  f16* Kc     = a16(LTB * HDIM);
  f16* Vc     = a16(LTB * HDIM);
  f16* outs   = a16(LTB * IDIM);     // per-step outputs (= next dec)
  f16* hPrev16 = a16((long)BSZ * HDIM);
  f16* hG16   = a16((long)BSZ * HDIM);
  f16* xa16   = a16((long)BSZ * HDIM);
  f16* qkv16  = a16((long)BSZ * 3 * HDIM);
  f16* probs  = a16((long)NHEADS * BSZ * BSZ);
  f16* attn16 = a16((long)BSZ * HDIM);
  f16* qc16   = a16((long)BSZ * HDIM);
  f16* s16    = a16((long)BSZ * HDIM);
  float* gi   = a32((long)BSZ * 3 * HDIM);
  float* gh   = a32((long)BSZ * 3 * HDIM);
  float* scores = a32((long)NHEADS * BSZ * BSZ);
  float* tmp  = a32((long)BSZ * HDIM);
  float* hPrev32 = a32((long)BSZ * HDIM);
  float* hG32 = a32((long)BSZ * HDIM);
  float* proj = a32(LTB * ODIM);

  auto cvt = [&](const float* s, f16* d, long n) {
    f32_to_f16_kernel<<<(int)((n + 255) / 256), 256, 0, stream>>>(s, d, n);
  };
  cvt(gru_Wih, wWih, (long)3 * HDIM * IDIM);
  cvt(gru_Whh, wWhh, (long)3 * HDIM * HDIM);
  cvt(enc_W, wEnc, (long)HDIM * IDIM);
  cvt(sa_in_w, wSaIn, (long)3 * HDIM * HDIM);
  cvt(sa_out_w, wSaOut, (long)HDIM * HDIM);
  cvt(sa_fc_w, wSaFc, (long)HDIM * HDIM);
  cvt(ca_in_w, wCaIn, (long)3 * HDIM * HDIM);
  cvt(ca_out_w, wCaOut, (long)HDIM * HDIM);
  cvt(fc_w, wFc, (long)IDIM * HDIM);
  cvt(outp_w, wOutp, (long)ODIM * IDIM);
  cvt(decoder_input, dec0, (long)BSZ * IDIM);

  // Normalize + pool -> pooled[l,t,b,:] (f16)
  pool_norm_kernel<<<LNUM * BSZ * TLEN, 256, 0, stream>>>(patch_features, pooled);

  // enc = ((pooled @ Wenc^T + b) @ Wenc^T + b)  (two chained GEMMs, M=10240)
  launch_gemm(stream, pooled, IDIM, 0, wEnc, IDIM, 0, 1, enc_b,
              nullptr, 0, 0, nullptr, encTmp, HDIM, 0, (int)LTB, HDIM, IDIM, 1);
  launch_gemm(stream, encTmp, HDIM, 0, wEnc, HDIM, 0, 1, enc_b,
              nullptr, 0, 0, nullptr, pooled, HDIM, 0, (int)LTB, HDIM, HDIM, 1);
  // Cross-attn K/V for all (l,t) slices
  launch_gemm(stream, pooled, HDIM, 0, wCaIn + (long)HDIM * HDIM, HDIM, 0, 1,
              ca_in_b + HDIM, nullptr, 0, 0, nullptr, Kc, HDIM, 0,
              (int)LTB, HDIM, HDIM, 1);
  launch_gemm(stream, pooled, HDIM, 0, wCaIn + (long)2 * HDIM * HDIM, HDIM, 0, 1,
              ca_in_b + 2 * HDIM, nullptr, 0, 0, nullptr, Vc, HDIM, 0,
              (int)LTB, HDIM, HDIM, 1);

  const float scale = 1.0f / sqrtf((float)HD);

  for (int l = 0; l < LNUM; ++l) {
    zero_state_kernel<<<(BSZ * HDIM + 255) / 256, 256, 0, stream>>>(
        hPrev32, hPrev16, (long)BSZ * HDIM);
    for (int t = 0; t < TLEN; ++t) {
      const f16* dec = (t == 0) ? dec0
                                : outs + (long)(l * TLEN + (t - 1)) * BSZ * IDIM;
      // --- GRU cell ---
      launch_gemm(stream, dec, IDIM, 0, wWih, IDIM, 0, 1, gru_bih,
                  nullptr, 0, 0, gi, nullptr, 3 * HDIM, 0, BSZ, 3 * HDIM, IDIM, 1);
      launch_gemm(stream, hPrev16, HDIM, 0, wWhh, HDIM, 0, 1, gru_bhh,
                  nullptr, 0, 0, gh, nullptr, 3 * HDIM, 0, BSZ, 3 * HDIM, HDIM, 1);
      gru_gates_kernel<<<(BSZ * HDIM + 255) / 256, 256, 0, stream>>>(
          gi, gh, hPrev32, hG32, hG16);
      // --- self-attention over batch dim (seq len 128, 8 heads, hd 96) ---
      launch_gemm(stream, hG16, HDIM, 0, wSaIn, HDIM, 0, 1, sa_in_b,
                  nullptr, 0, 0, nullptr, qkv16, 3 * HDIM, 0, BSZ, 3 * HDIM, HDIM, 1);
      launch_gemm(stream, qkv16, 3 * HDIM, HD, qkv16 + HDIM, 3 * HDIM, HD, 1,
                  nullptr, nullptr, 0, 0, scores, nullptr, BSZ, (long)BSZ * BSZ,
                  BSZ, BSZ, HD, NHEADS);
      softmax_rows_kernel<<<NHEADS * BSZ, 128, 0, stream>>>(scores, probs, scale);
      launch_gemm(stream, probs, BSZ, (long)BSZ * BSZ, qkv16 + 2 * HDIM, 3 * HDIM,
                  HD, 0, nullptr, nullptr, 0, 0, nullptr, attn16, HDIM, HD,
                  BSZ, HD, BSZ, NHEADS);
      launch_gemm(stream, attn16, HDIM, 0, wSaOut, HDIM, 0, 1, sa_out_b,
                  hG32, HDIM, 0, tmp, nullptr, HDIM, 0, BSZ, HDIM, HDIM, 1);
      layernorm_rows_kernel<<<BSZ, 256, 0, stream>>>(tmp, ln_g, ln_b,
                                                     nullptr, xa16);
      launch_gemm(stream, xa16, HDIM, 0, wSaFc, HDIM, 0, 1, sa_fc_b,
                  hG32, HDIM, 0, tmp, nullptr, HDIM, 0, BSZ, HDIM, HDIM, 1);
      // LN2 output is x AND the next-step GRU hidden state (carry = (out, x))
      layernorm_rows_kernel<<<BSZ, 256, 0, stream>>>(tmp, ln_g, ln_b,
                                                     hPrev32, hPrev16);
      // --- cross-attention vs enc slice (l,t) ---
      const f16* KcT = Kc + (long)(l * TLEN + t) * BSZ * HDIM;
      const f16* VcT = Vc + (long)(l * TLEN + t) * BSZ * HDIM;
      launch_gemm(stream, hPrev16, HDIM, 0, wCaIn, HDIM, 0, 1, ca_in_b,
                  nullptr, 0, 0, nullptr, qc16, HDIM, 0, BSZ, HDIM, HDIM, 1);
      launch_gemm(stream, qc16, HDIM, HD, KcT, HDIM, HD, 1, nullptr,
                  nullptr, 0, 0, scores, nullptr, BSZ, (long)BSZ * BSZ,
                  BSZ, BSZ, HD, NHEADS);
      softmax_rows_kernel<<<NHEADS * BSZ, 128, 0, stream>>>(scores, probs, scale);
      launch_gemm(stream, probs, BSZ, (long)BSZ * BSZ, VcT, HDIM, HD, 0, nullptr,
                  nullptr, 0, 0, nullptr, attn16, HDIM, HD, BSZ, HD, BSZ, NHEADS);
      // s = x + ctx  (residual fused into epilogue), f16 for next GEMM
      launch_gemm(stream, attn16, HDIM, 0, wCaOut, HDIM, 0, 1, ca_out_b,
                  hPrev32, HDIM, 0, nullptr, s16, HDIM, 0, BSZ, HDIM, HDIM, 1);
      // out = s @ fc_w^T + fc_b  -> outs[l,t] (also next step's dec)
      launch_gemm(stream, s16, HDIM, 0, wFc, HDIM, 0, 1, fc_b,
                  nullptr, 0, 0, nullptr,
                  outs + (long)(l * TLEN + t) * BSZ * IDIM, IDIM, 0,
                  BSZ, IDIM, HDIM, 1);
    }
  }

  // Output projection: [10240,768] @ [512,768]^T -> proj f32
  launch_gemm(stream, outs, IDIM, 0, wOutp, IDIM, 0, 1, outp_b,
              nullptr, 0, 0, proj, nullptr, ODIM, 0, (int)LTB, ODIM, IDIM, 1);
  // Fuse over L
  long total = (long)BSZ * TLEN * ODIM;
  fuse_out_kernel<<<(int)((total + 255) / 256), 256, 0, stream>>>(
      proj, fus_w, fus_b, (float*)d_out);
}
